// PolicyNet_10058813407476
// MI455X (gfx1250) — matrix operand order repaired and checked by reading, not compiled, not run on hardware
//
#include <hip/hip_runtime.h>
#include <hip/hip_bf16.h>

typedef __attribute__((ext_vector_type(8)))  _Float16 v8h;
typedef __attribute__((ext_vector_type(16))) _Float16 v16h;
typedef __attribute__((ext_vector_type(8)))  float    v8f;

// ---------------------------------------------------------------------------
// B-panel packing: element (k, n) -> panel[kb][nb][lane][half]
// lane 0-15 = column n%16 holding K kb*32+0..7,16..23 ; lanes 16-31 hold K +8..15,24..31
// ---------------------------------------------------------------------------
__device__ __forceinline__ size_t panel_idx(int k, int n, int NB) {
  int kb = k >> 5, kl = k & 31, nb = n >> 4, nl = n & 15;
  int lane = nl | (((kl >> 3) & 1) << 4);
  return ((((size_t)kb * NB + nb) * 32 + lane) << 4) + ((kl & 7) | ((kl >> 4) << 3));
}

__device__ __forceinline__ v16h make_afrag(v8h a0, v8h a1) {
  v16h a;
#pragma unroll
  for (int i = 0; i < 8; ++i) { a[i] = a0[i]; a[8 + i] = a1[i]; }
  return a;
}

// ---------------------------------------------------------------------------
// K1: conv1 (3->64, 3x3 SAME) + relu + maxpool2 ; out h1 f16 NHWC [256,16,16,64]
// ---------------------------------------------------------------------------
__global__ void conv1_pool_kernel(const float* __restrict__ x,
                                  const float* __restrict__ w,
                                  const float* __restrict__ bias,
                                  _Float16* __restrict__ h1) {
  int tid = blockIdx.x * 256 + threadIdx.x;      // 4,194,304 threads
  int c  = tid & 63;
  int p  = tid >> 6;
  int ow = p & 15; p >>= 4;
  int oh = p & 15;
  int b  = p >> 4;

  float wr[27];
#pragma unroll
  for (int i = 0; i < 27; ++i) wr[i] = w[c * 27 + i];
  float bv = bias[c];

  float best = 0.0f;   // relu output is >= 0
#pragma unroll
  for (int py = 0; py < 2; ++py)
#pragma unroll
  for (int px = 0; px < 2; ++px) {
    int y0 = oh * 2 + py, x0 = ow * 2 + px;
    float acc = bv;
    for (int ci = 0; ci < 3; ++ci)
#pragma unroll
      for (int dy = 0; dy < 3; ++dy) {
        int iy = y0 + dy - 1;
        if (iy < 0 || iy >= 32) continue;
#pragma unroll
        for (int dx = 0; dx < 3; ++dx) {
          int ix = x0 + dx - 1;
          if (ix < 0 || ix >= 32) continue;
          acc = fmaf(x[((b * 3 + ci) * 32 + iy) * 32 + ix], wr[(ci * 3 + dy) * 3 + dx], acc);
        }
      }
    best = fmaxf(best, acc > 0.0f ? acc : 0.0f);
  }
  h1[tid] = (_Float16)best;   // tid == ((b*16+oh)*16+ow)*64 + c  (NHWC)
}

// ---------------------------------------------------------------------------
// Pack kernels: f32 weights -> f16 WMMA B panels
// ---------------------------------------------------------------------------
__global__ void pack_conv2_w(const float* __restrict__ w, _Float16* __restrict__ bp) {
  int tid = blockIdx.x * 256 + threadIdx.x;      // 576*128
  int n = tid & 127, k = tid >> 7;
  int t = k >> 6, c = k & 63, dy = t / 3, dx = t % 3;     // k = tap*64 + cin
  bp[panel_idx(k, n, 8)] = (_Float16)w[((n * 64 + c) * 3 + dy) * 3 + dx];
}

__global__ void pack_fc_w(const float* __restrict__ w, _Float16* __restrict__ bp) {
  int tid = blockIdx.x * 256 + threadIdx.x;      // 8192*1024
  int n = tid & 1023, k = tid >> 10;
  int p = k >> 7, c = k & 127;                   // k = (y*8+x)*128 + c  (NHWC order)
  bp[panel_idx(k, n, 64)] = (_Float16)w[(size_t)n * 8192 + c * 64 + p]; // NCHW column
}

__global__ void pack_xy_w(const float* __restrict__ w, _Float16* __restrict__ bp) {
  int tid = blockIdx.x * 256 + threadIdx.x;      // 1280*512
  int n = tid & 511, k = tid >> 9;
  bp[panel_idx(k, n, 32)] = (_Float16)w[(size_t)n * 1280 + k];
}

// ---------------------------------------------------------------------------
// K3: conv2 implicit-GEMM via WMMA.
// Block = 8 waves = 8 m-tiles (m-tile = b*16+oh, tile rows = 16 ow positions).
// Each wave accumulates all 8 n-tiles (N=128), reusing its A fragment 8x.
// Per k-block the 8KB B slab is staged once into LDS and shared by all waves.
// All 8 B fragments are loaded into registers before the 8 back-to-back WMMAs
// (single s_wait_dscnt per k-block). k-loop fully unrolled (18 iterations) so
// the tap / padding arithmetic constant-folds.
// Fused relu; out f16 NHWC [256,16,16,128].
// ---------------------------------------------------------------------------
__global__ void conv2_gemm_kernel(const _Float16* __restrict__ h1,
                                  const _Float16* __restrict__ Bp,
                                  const float* __restrict__ bias,
                                  _Float16* __restrict__ out) {
  __shared__ __align__(32) _Float16 bs[4096];     // 8 ntiles * 32 lanes * 16 halfs
  int wave = threadIdx.x >> 5, lane = threadIdx.x & 31;
  int mtile = blockIdx.x * 8 + wave;              // grid.x = 512
  int b = mtile >> 4, oh = mtile & 15;
  int ow   = lane & 15;
  int koff = (lane >> 4) << 3;                    // A-frag: lanes>=16 hold K 8..15/24..31
  int t256 = threadIdx.x;

  v8f acc[8];
#pragma unroll
  for (int j = 0; j < 8; ++j) acc[j] = (v8f){};
  v8h zero = {};

#pragma unroll
  for (int kb = 0; kb < 18; ++kb) {
    __syncthreads();                              // previous slab fully consumed
    *(v16h*)(bs + t256 * 16) =
        *(const v16h*)(Bp + (size_t)kb * 4096 + t256 * 16);
    if (kb < 17)
      __builtin_prefetch(Bp + (size_t)(kb + 1) * 4096 + t256 * 16, 0, 1);

    // A fragment: on-the-fly im2col, branchless padding
    int t  = kb >> 1;                             // tap 0..8 (constant-folded)
    int c0 = ((kb & 1) << 5) + koff;              // input-channel base of chunk
    int dy = t / 3, dx = t % 3;
    int ih = oh + dy - 1, iw = ow + dx - 1;
    bool inb = (ih >= 0) && (ih < 16) && (iw >= 0) && (iw < 16);
    int ihc = ih < 0 ? 0 : (ih > 15 ? 15 : ih);
    int iwc = iw < 0 ? 0 : (iw > 15 ? 15 : iw);
    const _Float16* px = h1 + (((long)(b * 16 + ihc) * 16 + iwc) << 6) + c0;
    v8h a0 = *(const v8h*)(px);
    v8h a1 = *(const v8h*)(px + 16);
    a0 = inb ? a0 : zero;
    a1 = inb ? a1 : zero;
    v16h a = make_afrag(a0, a1);

    __syncthreads();                              // slab visible to all waves

    v16h bfr[8];                                  // gather all B frags first
#pragma unroll
    for (int j = 0; j < 8; ++j)
      bfr[j] = *(const v16h*)(bs + ((j * 32 + lane) << 4));
#pragma unroll
    for (int j = 0; j < 8; ++j)                   // then 8 back-to-back WMMAs
      acc[j] = __builtin_amdgcn_wmma_f32_16x16x32_f16(false, a, false, bfr[j],
                                                      (short)0, acc[j], false, false);
  }

  int nl  = lane & 15;
  int owb = (lane >> 4) << 3;                     // C/D layout: lanes>=16 hold M 8..15
  size_t base = ((size_t)mtile * 16 + owb) * 128;
#pragma unroll
  for (int j = 0; j < 8; ++j) {
    int c = j * 16 + nl;
    float bv = bias[c];
#pragma unroll
    for (int v = 0; v < 8; ++v) {
      float r = acc[j][v] + bv;
      r = r > 0.0f ? r : 0.0f;
      out[base + (size_t)v * 128 + c] = (_Float16)r;
    }
  }
}

// ---------------------------------------------------------------------------
// K4: maxpool2 on conv2 output -> h2 f16 NHWC [256,8,8,128]
// ---------------------------------------------------------------------------
__global__ void pool2_kernel(const _Float16* __restrict__ c2, _Float16* __restrict__ h2) {
  int tid = blockIdx.x * 256 + threadIdx.x;      // 2,097,152
  int c = tid & 127;
  int p = tid >> 7;
  int x8 = p & 7; p >>= 3;
  int y8 = p & 7;
  int b  = p >> 3;
  float m = -1e30f;
#pragma unroll
  for (int dy = 0; dy < 2; ++dy)
#pragma unroll
  for (int dx = 0; dx < 2; ++dx) {
    float v = (float)c2[(((size_t)(b * 16 + y8 * 2 + dy) * 16) + x8 * 2 + dx) * 128 + c];
    m = fmaxf(m, v);
  }
  h2[tid] = (_Float16)m;
}

// ---------------------------------------------------------------------------
// Generic wave-WMMA GEMM: C[M,N] = A[M,K](f16, row-major) * Bpanel + bias.
// Block = 8 waves = 8 m-tiles (grid.x = M/128); each wave accumulates the
// 4 n-tiles of group blockIdx.y, reusing A 4x. The 4KB B slab per k-block is
// staged in LDS once per block and shared by the 8 m-tile waves; all 4 B
// fragments are pulled to registers before the 4 back-to-back WMMAs.
// ---------------------------------------------------------------------------
__global__ void wgemm_f16_kernel(const _Float16* __restrict__ A, int K,
                                 const _Float16* __restrict__ Bp, int NB,
                                 const float* __restrict__ bias,
                                 float* __restrict__ C, int ldc) {
  __shared__ __align__(32) _Float16 bs[2048];     // 4 ntiles * 32 lanes * 16 halfs
  int wave = threadIdx.x >> 5, lane = threadIdx.x & 31;
  int mtile  = blockIdx.x * 8 + wave;
  int ntile0 = blockIdx.y * 4;
  int row  = mtile * 16 + (lane & 15);
  int koff = (lane >> 4) << 3;
  const _Float16* arow = A + (size_t)row * K + koff;
  int t256 = threadIdx.x;

  v8f acc[4];
#pragma unroll
  for (int j = 0; j < 4; ++j) acc[j] = (v8f){};

  int nkb = K >> 5;
  for (int kb = 0; kb < nkb; ++kb) {
    __syncthreads();
    *(v8h*)(bs + t256 * 8) =
        *(const v8h*)(Bp + (((size_t)kb * NB + ntile0) * 32) * 16 + t256 * 8);

    const _Float16* ap = arow + (kb << 5);
    v8h a0 = *(const v8h*)(ap);
    v8h a1 = *(const v8h*)(ap + 16);
    v16h a = make_afrag(a0, a1);
    if (kb + 1 < nkb) {
      __builtin_prefetch(ap + 32, 0, 1);
      __builtin_prefetch(Bp + (((size_t)(kb + 1) * NB + ntile0) * 32) * 16 + t256 * 8, 0, 1);
    }

    __syncthreads();

    v16h bfr[4];
#pragma unroll
    for (int j = 0; j < 4; ++j)
      bfr[j] = *(const v16h*)(bs + ((j * 32 + lane) << 4));
#pragma unroll
    for (int j = 0; j < 4; ++j)
      acc[j] = __builtin_amdgcn_wmma_f32_16x16x32_f16(false, a, false, bfr[j],
                                                      (short)0, acc[j], false, false);
  }

  int nl = lane & 15;
  int mb = mtile * 16 + ((lane >> 4) << 3);
#pragma unroll
  for (int j = 0; j < 4; ++j) {
    int n = (ntile0 + j) * 16 + nl;
    float bv = bias[n];
#pragma unroll
    for (int v = 0; v < 8; ++v) C[(size_t)(mb + v) * ldc + n] = acc[j][v] + bv;
  }
}

// ---------------------------------------------------------------------------
// K6: z2 = y @ enc_y_w.T + b  (K=100, VALU) -> z[:, 1024:1280]
// ---------------------------------------------------------------------------
__global__ void z2_kernel(const float* __restrict__ y, const float* __restrict__ w,
                          const float* __restrict__ bias, float* __restrict__ z) {
  int tid = blockIdx.x * 256 + threadIdx.x;      // 65536
  int n = tid & 255, b = tid >> 8;
  float acc = bias[n];
  for (int i = 0; i < 100; ++i) acc = fmaf(y[b * 100 + i], w[n * 100 + i], acc);
  z[(size_t)b * 1280 + 1024 + n] = acc;
}

// ---------------------------------------------------------------------------
// K8: L2 normalize each row of z [256,1280], relu, -> zn f16
// ---------------------------------------------------------------------------
__global__ void normalize_kernel(const float* __restrict__ z, _Float16* __restrict__ zn) {
  __shared__ float red[256];
  int b = blockIdx.x, t = threadIdx.x;
  const float* zr = z + (size_t)b * 1280;
  float ss = 0.0f;
  for (int i = t; i < 1280; i += 256) { float v = zr[i]; ss = fmaf(v, v, ss); }
  red[t] = ss; __syncthreads();
  for (int s = 128; s > 0; s >>= 1) { if (t < s) red[t] += red[t + s]; __syncthreads(); }
  float inv = 1.0f / fmaxf(sqrtf(red[0]), 1e-12f);
  _Float16* zo = zn + (size_t)b * 1280;
  for (int i = t; i < 1280; i += 256) {
    float v = zr[i] * inv;
    zo[i] = (_Float16)(v > 0.0f ? v : 0.0f);
  }
}

// ---------------------------------------------------------------------------
// K11: column mean of z3 [256,512] -> mvec [512]
// ---------------------------------------------------------------------------
__global__ void colmean_kernel(const float* __restrict__ z3, float* __restrict__ mvec) {
  int j = blockIdx.x * 256 + threadIdx.x;        // 512
  float s = 0.0f;
  for (int b = 0; b < 256; ++b) s += z3[(size_t)b * 512 + j];
  mvec[j] = s * (1.0f / 256.0f);
}

// ---------------------------------------------------------------------------
// K12: gumbel softmax (hard, straight-through fwd = one-hot bit) + softmax(512)
// ---------------------------------------------------------------------------
__global__ void final_kernel(const float* __restrict__ mvec,
                             const float* __restrict__ gumbel,
                             const int* __restrict__ temp,
                             float* __restrict__ out) {
  __shared__ float red[512];
  int j = threadIdx.x;                           // 512 threads
  float tau = (float)temp[0];
  float m  = mvec[j];
  float s0 = (-m + gumbel[2 * j + 0]) / tau;
  float s1 = ( m + gumbel[2 * j + 1]) / tau;
  float bit = (s1 >= s0) ? 1.0f : 0.0f;          // y_hard[:,1] forward value
  red[j] = bit; __syncthreads();
  for (int s = 256; s > 0; s >>= 1) { if (j < s) red[j] = fmaxf(red[j], red[j + s]); __syncthreads(); }
  float mx = red[0]; __syncthreads();
  float e = __expf(bit - mx);
  red[j] = e; __syncthreads();
  for (int s = 256; s > 0; s >>= 1) { if (j < s) red[j] += red[j + s]; __syncthreads(); }
  out[j] = e / red[0];
}

// ---------------------------------------------------------------------------
extern "C" void kernel_launch(void* const* d_in, const int* in_sizes, int n_in,
                              void* d_out, int out_size, void* d_ws, size_t ws_size,
                              hipStream_t stream) {
  const float* x        = (const float*)d_in[0];
  const float* y        = (const float*)d_in[1];
  const float* conv1_w  = (const float*)d_in[2];
  const float* conv1_b  = (const float*)d_in[3];
  const float* conv2_w  = (const float*)d_in[4];
  const float* conv2_b  = (const float*)d_in[5];
  const float* fc_x_w   = (const float*)d_in[6];
  const float* fc_x_b   = (const float*)d_in[7];
  const float* enc_y_w  = (const float*)d_in[8];
  const float* enc_y_b  = (const float*)d_in[9];
  const float* enc_xy_w = (const float*)d_in[10];
  const float* enc_xy_b = (const float*)d_in[11];
  const float* gumbel   = (const float*)d_in[12];
  const int*   temp     = (const int*)d_in[13];
  float* out = (float*)d_out;

  char* ws = (char*)d_ws;
  _Float16* h1      = (_Float16*)(ws + 0);            //  8,388,608 B
  _Float16* c2out   = (_Float16*)(ws + 8388608);      // 16,777,216 B (aliased w/ BpFC)
  _Float16* h2      = (_Float16*)(ws + 25165824);     //  4,194,304 B
  _Float16* BpC2    = (_Float16*)(ws + 29360128);     //    147,456 B
  _Float16* BpXY    = (_Float16*)(ws + 29507584);     //  1,310,720 B
  float*    z       = (float*)   (ws + 30818304);     //  1,310,720 B  [256,1280]
  _Float16* zn      = (_Float16*)(ws + 32129024);     //    655,360 B
  float*    z3      = (float*)   (ws + 32784384);     //    524,288 B  [256,512]
  float*    mvec    = (float*)   (ws + 33308672);     //      2,048 B
  _Float16* BpFC    = c2out;                          // reuse after pool2

  // 1) conv1 + relu + pool
  conv1_pool_kernel<<<16384, 256, 0, stream>>>(x, conv1_w, conv1_b, h1);
  // 2) pack conv2 weights
  pack_conv2_w<<<288, 256, 0, stream>>>(conv2_w, BpC2);
  // 3) conv2 implicit GEMM (WMMA, LDS-staged B) + relu
  conv2_gemm_kernel<<<512, 256, 0, stream>>>(h1, BpC2, conv2_b, c2out);
  // 4) maxpool2 -> h2 (= fc A-matrix, NHWC k-order)
  pool2_kernel<<<8192, 256, 0, stream>>>(c2out, h2);
  // 5) pack fc weights (into region previously holding c2out)
  pack_fc_w<<<32768, 256, 0, stream>>>(fc_x_w, BpFC);
  // 6) enc_y (small, VALU)
  z2_kernel<<<256, 256, 0, stream>>>(y, enc_y_w, enc_y_b, z);
  // 7) fc_x GEMM (WMMA): [256,8192]x[8192,1024] -> z[:, :1024]
  wgemm_f16_kernel<<<dim3(2, 16), 256, 0, stream>>>(h2, 8192, BpFC, 64, fc_x_b, z, 1280);
  // 8) L2 normalize + relu -> zn f16
  normalize_kernel<<<256, 256, 0, stream>>>(z, zn);
  // 9) pack enc_xy weights
  pack_xy_w<<<2560, 256, 0, stream>>>(enc_xy_w, BpXY);
  // 10) enc_xy GEMM (WMMA): [256,1280]x[1280,512] -> z3
  wgemm_f16_kernel<<<dim3(2, 8), 256, 0, stream>>>(zn, 1280, BpXY, 32, enc_xy_b, z3, 512);
  // 11) batch mean
  colmean_kernel<<<2, 256, 0, stream>>>(z3, mvec);
  // 12) gumbel / straight-through / softmax
  final_kernel<<<1, 512, 0, stream>>>(mvec, gumbel, temp, out);

  (void)in_sizes; (void)n_in; (void)out_size; (void)ws_size;
}
